// Gene_Linear_Attention_68229850465008
// MI455X (gfx1250) — matrix operand order repaired and checked by reading, not compiled, not run on hardware
//
#include <hip/hip_runtime.h>

#define DEV __device__ __forceinline__

typedef __attribute__((ext_vector_type(16))) __bf16 v16bf;
typedef __attribute__((ext_vector_type(8)))  __bf16 v8bf;
typedef __attribute__((ext_vector_type(4)))  __bf16 v4bf;
typedef __attribute__((ext_vector_type(8)))  float  v8f;

constexpr int R  = 64;
constexpr int C  = 2048;
constexpr int E  = 128;
constexpr int H  = 8;
constexpr int HD = 16;
constexpr int D  = R * HD;    // 1024
constexpr int LDSS = 40;      // padded LDS row stride (bf16 elems); 80B -> keeps 16B alignment

// ---------- helpers ----------

struct Frag32 { unsigned u[8]; };

// A fragment (16x32 bf16) from LDS tile stored [m][k], stride LDSS.
DEV v16bf load_a_frag(const __bf16* As, int m0, int lane) {
  int m  = m0 + (lane & 15);
  int kg = (lane >> 4) * 8;
  const unsigned* r32 = reinterpret_cast<const unsigned*>(As + m * LDSS);
  Frag32 t;
#pragma unroll
  for (int i = 0; i < 4; ++i) t.u[i]     = r32[(kg >> 1) + i];
#pragma unroll
  for (int i = 0; i < 4; ++i) t.u[4 + i] = r32[((16 + kg) >> 1) + i];
  return __builtin_bit_cast(v16bf, t);
}

// B fragment (32x16 bf16) from LDS tile stored transposed [n][k], stride LDSS.
DEV v16bf load_b_frag(const __bf16* Bs, int n0, int lane) {
  int n  = n0 + (lane & 15);
  int kb = (lane >> 4) * 8;
  const unsigned* r32 = reinterpret_cast<const unsigned*>(Bs + n * LDSS);
  Frag32 t;
#pragma unroll
  for (int i = 0; i < 4; ++i) t.u[i]     = r32[(kb >> 1) + i];
#pragma unroll
  for (int i = 0; i < 4; ++i) t.u[4 + i] = r32[((16 + kb) >> 1) + i];
  return __builtin_bit_cast(v16bf, t);
}

// Per-lane async DMA: 16B global -> LDS, tracked by ASYNCcnt (CDNA5 async path).
DEV void async_copy_b128(const __bf16* gaddr, __bf16* lds) {
  unsigned ldsoff = (unsigned)(uintptr_t)lds;   // flat LDS aperture: low 32 bits = LDS offset
  asm volatile("global_load_async_to_lds_b128 %0, %1, off"
               :: "v"(ldsoff), "v"(gaddr) : "memory");
}

DEV void wait_async_all() {
  asm volatile("s_wait_asynccnt 0x0" ::: "memory");
}

// f32 source tile (row-contiguous k) -> bf16 LDS tile [row][k]:
DEV void load_f32_tile_to_lds(__bf16* S, const float* src, int row0, int ld, int k0, int tid) {
#pragma unroll
  for (int j = 0; j < 4; ++j) {
    int chunk = tid + j * 256;
    int mm = chunk >> 3, part = chunk & 7;
    float4 v = *(const float4*)(src + (size_t)(row0 + mm) * ld + k0 + part * 4);
    v4bf o;
    o[0] = (__bf16)v.x; o[1] = (__bf16)v.y; o[2] = (__bf16)v.z; o[3] = (__bf16)v.w;
    *(v4bf*)(S + mm * LDSS + part * 4) = o;      // 8B LDS store (v_cvt_pk_bf16_f32 x2)
  }
}

// ---------- kernel 1: projection  y = x @ W^T + b, scattered to (H,C,D) ----------
template <bool BF16OUT>
__global__ __launch_bounds__(256) void proj_kernel(
    const float* __restrict__ x,     // (R*C, E) f32
    const float* __restrict__ W,     // (E, E)   f32
    const float* __restrict__ bias,  // (E)
    void* __restrict__ outp)         // (H, C, D) f32 or bf16
{
  __shared__ __bf16 As[128 * LDSS];
  __shared__ __bf16 Bs[128 * LDSS];
  const int tid  = threadIdx.x;
  const int lane = tid & 31;
  const int wave = tid >> 5;
  const int wm = wave >> 1, wn = wave & 1;   // 4 x 2 wave grid, 32x64 per wave
  const int m0 = blockIdx.x * 128;

  v8f acc[2][4];
#pragma unroll
  for (int tm = 0; tm < 2; ++tm)
#pragma unroll
    for (int tn = 0; tn < 4; ++tn) acc[tm][tn] = (v8f)0.0f;

  for (int k0 = 0; k0 < E; k0 += 32) {
    load_f32_tile_to_lds(As, x, m0, E, k0, tid);   // A'[m,k] = x[m*E+k]
    load_f32_tile_to_lds(Bs, W, 0,  E, k0, tid);   // B'[k,n] = W[n*E+k] -> stored [n][k]
    __syncthreads();

    v16bf af[2], bfr[4];
#pragma unroll
    for (int tm = 0; tm < 2; ++tm) af[tm] = load_a_frag(As, wm * 32 + tm * 16, lane);
#pragma unroll
    for (int tn = 0; tn < 4; ++tn) bfr[tn] = load_b_frag(Bs, wn * 64 + tn * 16, lane);
#pragma unroll
    for (int tm = 0; tm < 2; ++tm)
#pragma unroll
      for (int tn = 0; tn < 4; ++tn)
        acc[tm][tn] = __builtin_amdgcn_wmma_f32_16x16x32_bf16(
            false, af[tm], false, bfr[tn], (short)0, acc[tm][tn], false, false);
    __syncthreads();
  }

#pragma unroll
  for (int tm = 0; tm < 2; ++tm)
#pragma unroll
    for (int tn = 0; tn < 4; ++tn) {
      int col = wn * 64 + tn * 16 + (lane & 15);
      float bvs = bias[col];
      int h = col >> 4, hd = col & 15;
#pragma unroll
      for (int i = 0; i < 8; ++i) {
        int row = m0 + wm * 32 + tm * 16 + i + ((lane >> 4) * 8);
        int c = row & (C - 1);
        int r = row >> 11;                           // C = 2048
        float y = acc[tm][tn][i] + bvs;
        size_t dst = ((size_t)h * C + c) * D + (size_t)r * HD + hd;
        if (BF16OUT) ((__bf16*)outp)[dst] = (__bf16)y;
        else         ((float*)outp)[dst]  = y;
      }
    }
}

// ---------- kernel 2: q softmax over HD (16 contiguous) + scale 1/32 -> bf16 ----------
__global__ __launch_bounds__(256) void qsoftmax_kernel(const float* __restrict__ qp,
                                                       __bf16* __restrict__ qb) {
  size_t g = (size_t)blockIdx.x * blockDim.x + threadIdx.x;   // H*C*R groups
  const float4* p = (const float4*)(qp + g * 16);
  float vals[16];
#pragma unroll
  for (int w = 0; w < 4; ++w) {
    float4 v = p[w];
    vals[4 * w + 0] = v.x; vals[4 * w + 1] = v.y;
    vals[4 * w + 2] = v.z; vals[4 * w + 3] = v.w;
  }
  float m = vals[0];
#pragma unroll
  for (int i = 1; i < 16; ++i) m = fmaxf(m, vals[i]);
  float s = 0.f;
#pragma unroll
  for (int i = 0; i < 16; ++i) { vals[i] = __expf(vals[i] - m); s += vals[i]; }
  float inv = 0.03125f / s;                // softmax * dim^-0.5 (dim = 1024)
  v8bf a, b;
#pragma unroll
  for (int i = 0; i < 8; ++i) a[i] = (__bf16)(vals[i] * inv);
#pragma unroll
  for (int i = 0; i < 8; ++i) b[i] = (__bf16)(vals[8 + i] * inv);
  *(v8bf*)(qb + g * 16)     = a;
  *(v8bf*)(qb + g * 16 + 8) = b;
}

// ---------- kernel 3: k softmax over C, coalesced 3-pass -> bf16 ----------
__global__ __launch_bounds__(256) void ksoftmax_kernel(const float* __restrict__ kp,
                                                       __bf16* __restrict__ kb) {
  __shared__ float red[256];
  const int td = threadIdx.x & 31;
  const int tc = threadIdx.x >> 5;             // 0..7
  const int h  = blockIdx.x >> 5;              // 32 blocks per head (D/32)
  const int d0 = (blockIdx.x & 31) * 32;
  const float* base = kp + ((size_t)h * C) * D + d0 + td;

  float m = -1e30f;
  for (int c = tc; c < C; c += 8) m = fmaxf(m, base[(size_t)c * D]);
  red[threadIdx.x] = m; __syncthreads();
  if (tc == 0) {
    float mm = red[td];
#pragma unroll
    for (int j = 1; j < 8; ++j) mm = fmaxf(mm, red[j * 32 + td]);
    red[td] = mm;
  }
  __syncthreads();
  m = red[td];
  __syncthreads();

  float s = 0.f;
  for (int c = tc; c < C; c += 8) s += __expf(base[(size_t)c * D] - m);
  red[threadIdx.x] = s; __syncthreads();
  if (tc == 0) {
    float ss = red[td];
#pragma unroll
    for (int j = 1; j < 8; ++j) ss += red[j * 32 + td];
    red[td] = ss;
  }
  __syncthreads();
  float inv = 1.0f / red[td];

  __bf16* ob = kb + ((size_t)h * C) * D + d0 + td;
  for (int c = tc; c < C; c += 8)
    ob[(size_t)c * D] = (__bf16)(__expf(base[(size_t)c * D] - m) * inv);
}

// ---------- kernel 4: per-head bf16 WMMA GEMM, 256x128 block / 64x64 wave ----------
// Double-buffered LDS; non-transposed tiles use GLOBAL_LOAD_ASYNC_TO_LDS_B128 (ASYNCcnt),
// transposed tiles use manual load + ds_store transpose (DScnt, waited at barrier).
template <int TA, int TB, int EPI>
__global__ __launch_bounds__(256) void gemm_bf16(
    const __bf16* __restrict__ A, long long lda, long long Ah,
    const __bf16* __restrict__ B, long long ldb, long long Bh,
    void* __restrict__ Out, long long ldo, long long Oh,
    int M, int N, int K)
{
  __shared__ __bf16 As[2][256 * LDSS];   // 2 x 20 KB
  __shared__ __bf16 Bs[2][128 * LDSS];   // 2 x 10 KB
  const int tid  = threadIdx.x;
  const int lane = tid & 31;
  const int wave = tid >> 5;
  const int wm = wave >> 1, wn = wave & 1;       // 4 x 2 wave grid, 64x64 per wave
  const int m0 = blockIdx.y * 256;
  const int n0 = blockIdx.x * 128;
  const int h  = blockIdx.z;
  const __bf16* Ab = A + (size_t)h * Ah;
  const __bf16* Bb = B + (size_t)h * Bh;
  (void)M; (void)N;

  v8f acc[4][4];
#pragma unroll
  for (int tm = 0; tm < 4; ++tm)
#pragma unroll
    for (int tn = 0; tn < 4; ++tn) acc[tm][tn] = (v8f)0.0f;

  auto copy_issue = [&](int buf, int k0) {
    __bf16* Asb = &As[buf][0];
    __bf16* Bsb = &Bs[buf][0];
    // ---- A tile (256 x 32) -> As[m][k] ----
    if (TA == 0) {
#pragma unroll
      for (int j = 0; j < 4; ++j) {
        int chunk = tid + j * 256;                 // 1024 chunks of 8 bf16
        int mm = chunk >> 2, part = chunk & 3;
        async_copy_b128(Ab + (size_t)(m0 + mm) * lda + k0 + part * 8,
                        Asb + mm * LDSS + part * 8);
      }
    } else {
#pragma unroll
      for (int j = 0; j < 4; ++j) {
        int chunk = tid + j * 256;                 // 32 k-rows x 32 chunks, transpose
        int kk = chunk >> 5, part = chunk & 31;
        uint4 v = *(const uint4*)(Ab + (size_t)(k0 + kk) * lda + m0 + part * 8);
        const __bf16* e = (const __bf16*)&v;
#pragma unroll
        for (int q = 0; q < 8; ++q) Asb[(part * 8 + q) * LDSS + kk] = e[q];
      }
    }
    // ---- B tile (32 x 128) -> Bs[n][k] ----
    if (TB == 0) {
#pragma unroll
      for (int j = 0; j < 2; ++j) {
        int chunk = tid + j * 256;                 // 32 k-rows x 16 chunks, transpose
        int kk = chunk >> 4, part = chunk & 15;
        uint4 v = *(const uint4*)(Bb + (size_t)(k0 + kk) * ldb + n0 + part * 8);
        const __bf16* e = (const __bf16*)&v;
#pragma unroll
        for (int q = 0; q < 8; ++q) Bsb[(part * 8 + q) * LDSS + kk] = e[q];
      }
    } else {
#pragma unroll
      for (int j = 0; j < 2; ++j) {
        int chunk = tid + j * 256;                 // 128 n-rows x 4 chunks, direct
        int mm = chunk >> 2, part = chunk & 3;
        async_copy_b128(Bb + (size_t)(n0 + mm) * ldb + k0 + part * 8,
                        Bsb + mm * LDSS + part * 8);
      }
    }
  };

  copy_issue(0, 0);                                // prologue
  int buf = 0;
  for (int k0 = 0; k0 < K; k0 += 32, buf ^= 1) {
    wait_async_all();                              // own async DMAs for buf done
    __syncthreads();                               // all waves' copies visible; prev reads done
    if (k0 + 32 < K) copy_issue(buf ^ 1, k0 + 32); // background DMA next tile

    const __bf16* Asb = &As[buf][0];
    const __bf16* Bsb = &Bs[buf][0];
    v16bf af[4], bfr[4];
#pragma unroll
    for (int tm = 0; tm < 4; ++tm) af[tm] = load_a_frag(Asb, wm * 64 + tm * 16, lane);
#pragma unroll
    for (int tn = 0; tn < 4; ++tn) bfr[tn] = load_b_frag(Bsb, wn * 64 + tn * 16, lane);
#pragma unroll
    for (int tm = 0; tm < 4; ++tm)
#pragma unroll
      for (int tn = 0; tn < 4; ++tn)
        acc[tm][tn] = __builtin_amdgcn_wmma_f32_16x16x32_bf16(
            false, af[tm], false, bfr[tn], (short)0, acc[tm][tn], false, false);
  }

#pragma unroll
  for (int tm = 0; tm < 4; ++tm)
#pragma unroll
    for (int tn = 0; tn < 4; ++tn) {
      int col = n0 + wn * 64 + tn * 16 + (lane & 15);
#pragma unroll
      for (int i = 0; i < 8; ++i) {
        int row = m0 + wm * 64 + tm * 16 + i + ((lane >> 4) * 8);
        float v = acc[tm][tn][i];
        if (EPI == 0) {
          ((float*)Out)[(size_t)h * Oh + (size_t)row * ldo + col] = v;
        } else if (EPI == 1) {
          ((__bf16*)Out)[(size_t)h * Oh + (size_t)row * ldo + col] = (__bf16)v;
        } else {
          int r = col >> 4, hd = col & 15;          // col = d = r*16+hd, row = c
          ((float*)Out)[((size_t)r * C + row) * E + (size_t)h * HD + hd] = v;
        }
      }
    }
}

// ---------- launcher ----------
extern "C" void kernel_launch(void* const* d_in, const int* in_sizes, int n_in,
                              void* d_out, int out_size, void* d_ws, size_t ws_size,
                              hipStream_t stream) {
  const float* x  = (const float*)d_in[0];
  const float* Wq = (const float*)d_in[1];
  const float* bq = (const float*)d_in[2];
  const float* Wk = (const float*)d_in[3];
  const float* bk = (const float*)d_in[4];
  const float* Wv = (const float*)d_in[5];
  const float* bv = (const float*)d_in[6];
  (void)in_sizes; (void)n_in; (void)out_size; (void)ws_size;

  float* out  = (float*)d_out;                          // (R,C,E)
  float* attn = (float*)d_out + (size_t)R * C * E;      // (H,C,C)

  char* ws = (char*)d_ws;
  const size_t szHCDf = (size_t)H * C * D * sizeof(float);   // 64 MB
  const size_t szHCDb = (size_t)H * C * D * sizeof(__bf16);  // 32 MB
  float*  q_pre = (float*)(ws);
  float*  k_pre = (float*)(ws + szHCDf);
  __bf16* qb    = (__bf16*)(ws + 2 * szHCDf);
  __bf16* kb    = (__bf16*)(ws + 2 * szHCDf + szHCDb);
  __bf16* vb    = (__bf16*)(ws + 2 * szHCDf + 2 * szHCDb);
  __bf16* ctx   = (__bf16*)(ws + 2 * szHCDf + 3 * szHCDb);   // (H,D,D) bf16

  dim3 blk(256);
  proj_kernel<false><<<dim3((R * C) / 128), blk, 0, stream>>>(x, Wq, bq, (void*)q_pre);
  proj_kernel<false><<<dim3((R * C) / 128), blk, 0, stream>>>(x, Wk, bk, (void*)k_pre);
  proj_kernel<true ><<<dim3((R * C) / 128), blk, 0, stream>>>(x, Wv, bv, (void*)vb);
  qsoftmax_kernel<<<dim3((H * C * R) / 256), blk, 0, stream>>>(q_pre, qb);
  ksoftmax_kernel<<<dim3((H * D) / 32), blk, 0, stream>>>(k_pre, kb);

  // attn[h] = q_scaled (CxD) @ k^T (DxC)   -> f32 (H,C,C)   [A async, B async]
  gemm_bf16<0, 1, 0><<<dim3(C / 128, C / 256, H), blk, 0, stream>>>(
      qb, (long long)D, (long long)C * D,
      kb, (long long)D, (long long)C * D,
      (void*)attn, (long long)C, (long long)C * C, C, C, D);
  // ctx[h] = k^T (DxC) @ v (CxD)           -> bf16 (H,D,D)  [manual-T both]
  gemm_bf16<1, 0, 1><<<dim3(D / 128, D / 256, H), blk, 0, stream>>>(
      kb, (long long)D, (long long)C * D,
      vb, (long long)D, (long long)C * D,
      (void*)ctx, (long long)D, (long long)D * D, D, D, C);
  // out[h] = q_scaled (CxD) @ ctx (DxD)    -> scattered f32 (R,C,E) [A async, B manual-T]
  gemm_bf16<0, 0, 2><<<dim3(D / 128, C / 256, H), blk, 0, stream>>>(
      qb, (long long)D, (long long)C * D,
      ctx, (long long)D, (long long)D * D,
      (void*)out, 0, 0, C, D, D);
}
